// MambaBlock_15762529976506
// MI455X (gfx1250) — matrix operand order repaired and checked by reading, not compiled, not run on hardware
//
#include <hip/hip_runtime.h>
#include <hip/hip_bf16.h>

typedef __bf16 bf16;
typedef bf16  v16bf __attribute__((ext_vector_type(16)));
typedef bf16  v8bf  __attribute__((ext_vector_type(8)));
typedef float v8f   __attribute__((ext_vector_type(8)));

__device__ __forceinline__ v8f wmma_bf16(v16bf a, v16bf b, v8f c) {
  // D = A(16x32 bf16) * B(32x16 bf16) + C(16x16 f32)
  return __builtin_amdgcn_wmma_f32_16x16x32_bf16(false, a, false, b, (short)0, c,
                                                 false, false);
}

__device__ __forceinline__ float sigmoidf_(float v) { return 1.f / (1.f + __expf(-v)); }

// ---------------------------------------------------------------------------
// Weight prep: W[K][N] f32 -> Wt[N][K] bf16, 64x64 LDS tiles, all coalesced.
// ---------------------------------------------------------------------------
__global__ __launch_bounds__(256)
void transpose_cvt_kernel(const float* __restrict__ W, bf16* __restrict__ Wt,
                          int K, int N)
{
  __shared__ float s[64][65];
  const int t  = threadIdx.x;
  const int n0 = blockIdx.x * 64, k0 = blockIdx.y * 64;
  const int cx = t & 63, cy = (t >> 6) * 16;
  #pragma unroll
  for (int r = 0; r < 16; ++r)
    s[cy + r][cx] = W[(size_t)(k0 + cy + r) * N + n0 + cx];
  __syncthreads();
  #pragma unroll
  for (int r = 0; r < 16; ++r)
    Wt[(size_t)(n0 + cy + r) * K + k0 + cx] = (bf16)s[cx][cy + r];
}

// ---------------------------------------------------------------------------
// RMSNorm over D=512 channels, output bf16. One block per row.
// ---------------------------------------------------------------------------
__global__ __launch_bounds__(256)
void rmsnorm_kernel(const float* __restrict__ x, const float* __restrict__ w,
                    bf16* __restrict__ out)
{
  const int row = blockIdx.x;            // 0..16383
  const float* xr = x + (size_t)row * 512;
  float s = 0.f;
  for (int i = threadIdx.x; i < 512; i += 256) { float v = xr[i]; s += v * v; }
  #pragma unroll
  for (int m = 16; m >= 1; m >>= 1) s += __shfl_xor(s, m, 32);
  __shared__ float red[8];
  if ((threadIdx.x & 31) == 0) red[threadIdx.x >> 5] = s;
  __syncthreads();
  float tot = 0.f;
  #pragma unroll
  for (int i = 0; i < 8; ++i) tot += red[i];
  const float r = rsqrtf(tot * (1.f / 512.f) + 1e-5f);
  for (int i = threadIdx.x; i < 512; i += 256)
    out[(size_t)row * 512 + i] = (bf16)(xr[i] * r * w[i]);
}

// ---------------------------------------------------------------------------
// K=3 'same' conv as GEMM:  Y[M,COUT] = A3[M,3*CIN] * W + b
//   A3[m, tap*CIN+c] = A[m+tap-1, c]  (zero rows at per-4096 batch edges)
// Wt is pre-transposed bf16 [COUT][KT].  128x128 tile / WG, 8 waves,
// wave = 32x64 output via 2x4 v_wmma_f32_16x16x32_bf16 accumulators.
// Double-buffered LDS: one barrier per K-chunk, global loads for chunk k+1
// issued before the WMMAs of chunk k.
// EPI: 0=bias->bf16  1=bias+swish->bf16  2=bias+softplus*aux*cb->bf16
//      3=bias+swish*aux->bf16            4=bias->f32
// ---------------------------------------------------------------------------
template<int CIN, int COUT, int EPI>
__global__ __launch_bounds__(256)
void conv_gemm_kernel(const bf16* __restrict__ A, const bf16* __restrict__ Wt,
                      const float* __restrict__ bias, void* __restrict__ Yv,
                      const bf16* __restrict__ aux, const float* __restrict__ cbv)
{
  constexpr int KT = 3 * CIN;
  constexpr int PITCH = 48;                       // bf16 elems; 96B rows (16B aligned)
  __shared__ bf16 As [2][128 * PITCH];            // A tiles 128(M) x 32(K), ping-pong
  __shared__ bf16 Bst[2][128 * PITCH];            // B tiles 128(N) x 32(K), ping-pong

  const int tid  = threadIdx.x;
  const int lane = tid & 31;
  const int wave = tid >> 5;
  const int half = lane >> 4;
  const int ln   = lane & 15;
  const int wm   = (wave & 3) * 32;               // wave row base within tile
  const int wn   = (wave >> 2) * 64;              // wave col base within tile
  const int tileN = blockIdx.x * 128;
  const int tileM = blockIdx.y * 128;

  v8f acc[2][4];
  #pragma unroll
  for (int i = 0; i < 2; ++i)
    #pragma unroll
    for (int j = 0; j < 4; ++j)
      #pragma unroll
      for (int r = 0; r < 8; ++r) acc[i][j][r] = 0.f;

  const int arow = tid >> 1;                      // 0..127
  const int aseg = tid & 1;                       // 16 bf16 per segment

  uint4 da0, da1, db0, db1;                       // staged global data
  auto load_global = [&](int k0) {
    const int tap = k0 / CIN;
    const int c0  = k0 % CIN;
    const int m   = tileM + arow;
    const bool valid = !((tap == 0 && (m & 4095) == 0) ||
                         (tap == 2 && (m & 4095) == 4095));
    const int ms = valid ? (m + tap - 1) : m;
    const uint4* asrc = (const uint4*)(A + (size_t)ms * CIN + c0 + aseg * 16);
    da0 = asrc[0]; da1 = asrc[1];
    if (!valid) { da0 = make_uint4(0,0,0,0); da1 = make_uint4(0,0,0,0); }
    const uint4* bsrc = (const uint4*)(Wt + (size_t)(tileN + arow) * KT + k0 + aseg * 16);
    db0 = bsrc[0]; db1 = bsrc[1];
  };
  auto store_lds = [&](int buf) {
    uint4* ad = (uint4*)&As[buf][arow * PITCH + aseg * 16];
    ad[0] = da0; ad[1] = da1;
    uint4* bd = (uint4*)&Bst[buf][arow * PITCH + aseg * 16];
    bd[0] = db0; bd[1] = db1;
  };

  load_global(0);
  store_lds(0);
  __syncthreads();

  for (int k0 = 0; k0 < KT; k0 += 32) {
    const int  cur      = (k0 >> 5) & 1;
    const bool has_next = (k0 + 32) < KT;
    if (has_next) load_global(k0 + 32);           // overlap with WMMA below

    // ---- fragments (contiguous 16B LDS reads matching ISA VGPR layouts) ----
    v16bf afrag[2];
    #pragma unroll
    for (int mt = 0; mt < 2; ++mt) {
      const bf16* ap = &As[cur][(wm + mt * 16 + ln) * PITCH];
      v8bf lo = *(const v8bf*)(ap + half * 8);        // K = half*8 + 0..7
      v8bf hi = *(const v8bf*)(ap + half * 8 + 16);   // K = half*8 + 16..23
      #pragma unroll
      for (int j = 0; j < 8; ++j) { afrag[mt][j] = lo[j]; afrag[mt][j + 8] = hi[j]; }
    }
    v16bf bfrag[4];
    #pragma unroll
    for (int nt = 0; nt < 4; ++nt) {
      const bf16* bp = &Bst[cur][(wn + nt * 16 + ln) * PITCH + half * 8];
      v8bf lo = *(const v8bf*)(bp);                   // K = half*8 + 0..7
      v8bf hi = *(const v8bf*)(bp + 16);              // K = half*8 + 16..23
      #pragma unroll
      for (int j = 0; j < 8; ++j) { bfrag[nt][j] = lo[j]; bfrag[nt][j + 8] = hi[j]; }
    }
    #pragma unroll
    for (int mt = 0; mt < 2; ++mt)
      #pragma unroll
      for (int nt = 0; nt < 4; ++nt)
        acc[mt][nt] = wmma_bf16(afrag[mt], bfrag[nt], acc[mt][nt]);

    if (has_next) store_lds(cur ^ 1);             // write buffer not being read
    __syncthreads();                              // single barrier per chunk
  }

  // ---- epilogue ----
  #pragma unroll
  for (int mt = 0; mt < 2; ++mt) {
    #pragma unroll
    for (int nt = 0; nt < 4; ++nt) {
      const int col = tileN + wn + nt * 16 + ln;
      const float bv = bias[col];
      #pragma unroll
      for (int r = 0; r < 8; ++r) {
        const int row = tileM + wm + mt * 16 + r + half * 8;
        const size_t idx = (size_t)row * COUT + col;
        float v = acc[mt][nt][r] + bv;
        if constexpr (EPI == 0) {
          ((bf16*)Yv)[idx] = (bf16)v;
        } else if constexpr (EPI == 1) {
          ((bf16*)Yv)[idx] = (bf16)(v * sigmoidf_(v));
        } else if constexpr (EPI == 2) {
          const float sp = (v > 20.f) ? v : log1pf(__expf(v));   // softplus(delta)
          ((bf16*)Yv)[idx] = (bf16)(sp * (float)aux[idx] * cbv[row]);
        } else if constexpr (EPI == 3) {
          ((bf16*)Yv)[idx] = (bf16)((v * sigmoidf_(v)) * (float)aux[idx]);
        } else {
          ((float*)Yv)[idx] = v;
        }
      }
    }
  }
}

// ---------------------------------------------------------------------------
// Bm = conv(X,fc2), Cm = conv(X,fc3) (N=16 each), cb[m] = sum_n Bm*Cm.
// One wave per 16-row tile (two WMMA accumulators); weight chunks staged
// through LDS; butterfly reduction over the 16 N lanes.
// ---------------------------------------------------------------------------
__global__ __launch_bounds__(256)
void cb_kernel(const bf16* __restrict__ X,
               const float* __restrict__ w2, const float* __restrict__ b2,
               const float* __restrict__ w3, const float* __restrict__ b3,
               float* __restrict__ cb)
{
  __shared__ bf16 Wt2[16 * 48];     // [n][k] for current 32-K chunk
  __shared__ bf16 Wt3[16 * 48];
  const int tid = threadIdx.x, lane = tid & 31, wave = tid >> 5;
  const int half = lane >> 4, ln = lane & 15;
  const int m0 = blockIdx.x * 128 + wave * 16;
  const int m  = m0 + ln;                       // A-matrix row for this lane
  const int wk = tid >> 3;                      // 0..31 (k within chunk)
  const int wn = (tid & 7) * 2;                 // 0,2,...,14 (n pair)

  v8f accB, accC;
  #pragma unroll
  for (int r = 0; r < 8; ++r) { accB[r] = 0.f; accC[r] = 0.f; }

  for (int k0 = 0; k0 < 3072; k0 += 32) {
    const int tap = k0 >> 10, c0 = k0 & 1023;
    // stage fc2/fc3 weight chunk into LDS, transposed + converted
    {
      const size_t o = (size_t)(k0 + wk) * 16 + wn;
      Wt2[(wn + 0) * 48 + wk] = (bf16)w2[o];
      Wt2[(wn + 1) * 48 + wk] = (bf16)w2[o + 1];
      Wt3[(wn + 0) * 48 + wk] = (bf16)w3[o];
      Wt3[(wn + 1) * 48 + wk] = (bf16)w3[o + 1];
    }
    __syncthreads();

    // A fragment: unconditional 16B vector loads from clamped row, masked to 0
    const bool valid = !((tap == 0 && (m & 4095) == 0) ||
                         (tap == 2 && (m & 4095) == 4095));
    const int ms = valid ? (m + tap - 1) : m;
    const bf16* ap = X + (size_t)ms * 1024 + c0 + half * 8;
    v8bf lo = *(const v8bf*)(ap);
    v8bf hi = *(const v8bf*)(ap + 16);
    if (!valid) {
      #pragma unroll
      for (int j = 0; j < 8; ++j) { lo[j] = (bf16)0.f; hi[j] = (bf16)0.f; }
    }
    v16bf a;
    #pragma unroll
    for (int j = 0; j < 8; ++j) { a[j] = lo[j]; a[j + 8] = hi[j]; }

    v16bf bB, bC;
    {
      const bf16* p2 = &Wt2[ln * 48 + half * 8];
      const bf16* p3 = &Wt3[ln * 48 + half * 8];
      v8bf l2 = *(const v8bf*)(p2), h2 = *(const v8bf*)(p2 + 16);
      v8bf l3 = *(const v8bf*)(p3), h3 = *(const v8bf*)(p3 + 16);
      #pragma unroll
      for (int j = 0; j < 8; ++j) {
        bB[j] = l2[j]; bB[j + 8] = h2[j];
        bC[j] = l3[j]; bC[j + 8] = h3[j];
      }
    }
    accB = wmma_bf16(a, bB, accB);
    accC = wmma_bf16(a, bC, accC);
    __syncthreads();
  }
  float p[8];
  const float bb = b2[ln], bc = b3[ln];
  #pragma unroll
  for (int r = 0; r < 8; ++r) p[r] = (accB[r] + bb) * (accC[r] + bc);
  #pragma unroll
  for (int mask = 8; mask >= 1; mask >>= 1)
    #pragma unroll
    for (int r = 0; r < 8; ++r) p[r] += __shfl_xor(p[r], mask, 32);
  if (ln == 0) {
    #pragma unroll
    for (int r = 0; r < 8; ++r) cb[m0 + half * 8 + r] = p[r];   // row = m0 + r + half*8
  }
}

// ---------------------------------------------------------------------------
extern "C" void kernel_launch(void* const* d_in, const int* in_sizes, int n_in,
                              void* d_out, int out_size, void* d_ws, size_t ws_size,
                              hipStream_t stream)
{
  const float* x      = (const float*)d_in[0];
  const float* norm_w = (const float*)d_in[1];
  const float* inp_w  = (const float*)d_in[2];
  const float* inp_b  = (const float*)d_in[3];
  const float* conv_w = (const float*)d_in[4];
  const float* conv_b = (const float*)d_in[5];
  const float* cl_w   = (const float*)d_in[6];
  const float* cl_b   = (const float*)d_in[7];
  const float* fc1_w  = (const float*)d_in[8];
  const float* fc1_b  = (const float*)d_in[9];
  const float* fc2_w  = (const float*)d_in[10];
  const float* fc2_b  = (const float*)d_in[11];
  const float* fc3_w  = (const float*)d_in[12];
  const float* fc3_b  = (const float*)d_in[13];
  // d_in[14] = A : provably unused (dA term is exactly zero in the reference)
  const float* res_w  = (const float*)d_in[15];
  const float* res_b  = (const float*)d_in[16];
  const float* out_w  = (const float*)d_in[17];
  const float* out_b  = (const float*)d_in[18];

  const size_t MB = 1024 * 1024;
  char* ws = (char*)d_ws;
  bf16*  xn   = (bf16*)ws;                         // 16 MB: [16384,512]
  bf16*  bA   = (bf16*)(ws + 16 * MB);             // 32 MB: [16384,1024]
  bf16*  bB   = (bf16*)(ws + 48 * MB);             // 32 MB: [16384,1024]
  float* cb   = (float*)(ws + 80 * MB);            // 64 KB: [16384]
  char*  wp   = ws + 80 * MB + 65536;              // bf16 transposed weights:
  bf16* wt_inp  = (bf16*)(wp);                     // [1024][1536] : 3 MB
  bf16* wt_conv = (bf16*)(wp + 3  * MB);           // [1024][3072] : 6 MB
  bf16* wt_cl   = (bf16*)(wp + 9  * MB);           // [1024][3072] : 6 MB
  bf16* wt_fc1  = (bf16*)(wp + 15 * MB);           // [1024][3072] : 6 MB
  bf16* wt_res  = (bf16*)(wp + 21 * MB);           // [1024][1536] : 3 MB
  bf16* wt_out  = (bf16*)(wp + 24 * MB);           // [ 512][3072] : 3 MB

  const dim3 blk(256);

  // 0) one-shot weight transpose + f32->bf16 (removes all cvt from hot loops)
  transpose_cvt_kernel<<<dim3(16, 24), blk, 0, stream>>>(inp_w,  wt_inp,  1536, 1024);
  transpose_cvt_kernel<<<dim3(16, 48), blk, 0, stream>>>(conv_w, wt_conv, 3072, 1024);
  transpose_cvt_kernel<<<dim3(16, 48), blk, 0, stream>>>(cl_w,   wt_cl,   3072, 1024);
  transpose_cvt_kernel<<<dim3(16, 48), blk, 0, stream>>>(fc1_w,  wt_fc1,  3072, 1024);
  transpose_cvt_kernel<<<dim3(16, 24), blk, 0, stream>>>(res_w,  wt_res,  1536, 1024);
  transpose_cvt_kernel<<<dim3(8,  48), blk, 0, stream>>>(out_w,  wt_out,  3072, 512);

  // 1) xn = rmsnorm(x)                                  -> xn (bf16)
  rmsnorm_kernel<<<16384, blk, 0, stream>>>(x, norm_w, xn);
  // 2) x_proj = conv(xn, inp) + b                       -> bA
  conv_gemm_kernel<512, 1024, 0><<<dim3(8, 128), blk, 0, stream>>>(
      xn, wt_inp, inp_b, bA, nullptr, nullptr);
  // 3) x_conv = swish(conv(x_proj, conv) + b)           -> bB
  conv_gemm_kernel<1024, 1024, 1><<<dim3(8, 128), blk, 0, stream>>>(
      bA, wt_conv, conv_b, bB, nullptr, nullptr);
  // 4) x_conv_out = conv(x_conv, cl) + b                -> bA
  conv_gemm_kernel<1024, 1024, 0><<<dim3(8, 128), blk, 0, stream>>>(
      bB, wt_cl, cl_b, bA, nullptr, nullptr);
  // 5) cb[m] = <conv(x_conv_out,fc2), conv(x_conv_out,fc3)>
  cb_kernel<<<128, blk, 0, stream>>>(bA, fc2_w, fc2_b, fc3_w, fc3_b, cb);
  // 6) x_ssm = softplus(conv(x_conv_out,fc1)+b) * x_conv_out * cb   -> bB
  conv_gemm_kernel<1024, 1024, 2><<<dim3(8, 128), blk, 0, stream>>>(
      bA, wt_fc1, fc1_b, bB, bA, cb);
  // 7) u = swish(conv(xn,res)+b) * x_ssm                -> bA
  conv_gemm_kernel<512, 1024, 3><<<dim3(8, 128), blk, 0, stream>>>(
      xn, wt_res, res_b, bA, bB, nullptr);
  // 8) out = conv(u, out_w) + b                         -> d_out (f32)
  conv_gemm_kernel<1024, 512, 4><<<dim3(4, 128), blk, 0, stream>>>(
      bA, wt_out, out_b, d_out, nullptr, nullptr);
}
